// SelfAttentionRPR_19078244729547
// MI455X (gfx1250) — matrix operand order repaired and checked by reading, not compiled, not run on hardware
//
#include <hip/hip_runtime.h>

#define TB 8
#define TT 1024
#define TD 128
#define RK 64         // relative window K
#define NPE 129       // 2K+1
#define PEK_N 144     // 129 padded to 16
#define PEV_K 160     // 129 padded to 32

typedef __attribute__((ext_vector_type(16))) _Float16 v16h;
typedef __attribute__((ext_vector_type(8)))  _Float16 h8;
typedef __attribute__((ext_vector_type(8)))  float    v8f;

// ---- WMMA f16 (A 16x32, B 32x16, C/D f32 16x16), wave32 ----
static __device__ __forceinline__ v8f wmma_f16(v16h a, v16h b, v8f c) {
  return __builtin_amdgcn_wmma_f32_16x16x32_f16(false, a, false, b, (short)0, c,
                                                false, false);
}

// A-matrix 16x32 f16 tile load from row-major [16 x >=32] source (stride ldr halves).
// ISA layout: lanes0-15 row M=lane, K: v0..3 -> 0..7, v4..7 -> 16..23; lanes16-31: +8.
static __device__ __forceinline__ v16h ld_a16(const _Float16* base, int ldr) {
  int lane = threadIdx.x & 31;
  const _Float16* p = base + (size_t)(lane & 15) * ldr + ((lane >> 4) << 3);
  h8 lo = *(const h8*)(p);
  h8 hi = *(const h8*)(p + 16);
  v16h r;
#pragma unroll
  for (int e = 0; e < 8; ++e) { r[e] = lo[e]; r[e + 8] = hi[e]; }
  return r;
}

// B-matrix 32x16 f16 tile: source rows = output columns n (row-major [16 x >=32], stride ldr).
// ISA layout: lanes0-15 row N=lane, K: v0..3 -> 0..7, v4..7 -> 16..23; lanes16-31: +8.
static __device__ __forceinline__ v16h ld_b16(const _Float16* base, int ldr) {
  int lane = threadIdx.x & 31;
  const _Float16* p = base + (size_t)(lane & 15) * ldr + ((lane >> 4) << 3);
  h8 lo = *(const h8*)(p);
  h8 hi = *(const h8*)(p + 16);
  v16h r;
#pragma unroll
  for (int e = 0; e < 8; ++e) { r[e] = lo[e]; r[e + 8] = hi[e]; }
  return r;
}

// ---------------- prep kernels ----------------
__global__ void cvt_f16_kernel(const float* __restrict__ src, _Float16* __restrict__ dst, int n) {
  int i = blockIdx.x * blockDim.x + threadIdx.x;
  if (i < n) dst[i] = (_Float16)src[i];
}
__global__ void transpose_w_kernel(const float* __restrict__ W, _Float16* __restrict__ WT) {
  int i = blockIdx.x * blockDim.x + threadIdx.x;   // 128*128
  int n = i >> 7, k = i & 127;
  WT[n * TD + k] = (_Float16)W[k * TD + n];
}
__global__ void prep_pek_kernel(const float* __restrict__ pek, _Float16* __restrict__ dst) {
  int i = blockIdx.x * blockDim.x + threadIdx.x;   // PEK_N*TD
  if (i >= PEK_N * TD) return;
  int r = i >> 7, d = i & 127;
  dst[i] = (_Float16)(r < NPE ? pek[r * TD + d] : 0.0f);
}
__global__ void prep_pev_kernel(const float* __restrict__ pev, _Float16* __restrict__ dst) {
  int i = blockIdx.x * blockDim.x + threadIdx.x;   // TD*PEV_K
  if (i >= TD * PEV_K) return;
  int d = i / PEV_K, r = i % PEV_K;
  dst[i] = (_Float16)(r < NPE ? pev[r * TD + d] : 0.0f);
}

// ---------------- generic WMMA GEMM: Y = act(A[MxK] @ W + bias), W given as WT[N x K] f16 --------
// act: 0=none 1=relu 2=tanh. Optional outputs: f32 (stride ldYf), f16 (stride N), f16 transposed (stride M).
__global__ void gemm16_kernel(const _Float16* __restrict__ A, const _Float16* __restrict__ BT,
                              const float* __restrict__ bias,
                              float* __restrict__ Yf, _Float16* __restrict__ Yh,
                              _Float16* __restrict__ YhT,
                              int M, int N, int Kd, int act, int ldYf) {
  int tile = blockIdx.x * (blockDim.x >> 5) + (threadIdx.x >> 5);
  int tn = N >> 4;
  int ntiles = (M >> 4) * tn;
  if (tile >= ntiles) return;                 // wave-uniform exit; EXEC stays full for live waves
  int mt = tile / tn, nt = tile % tn;
  int lane = threadIdx.x & 31;
  v8f acc = {};
  for (int kk = 0; kk < Kd; kk += 32) {
    v16h a = ld_a16(A + (size_t)mt * 16 * Kd + kk, Kd);
    v16h b = ld_b16(BT + (size_t)nt * 16 * Kd + kk, Kd);
    acc = wmma_f16(a, b, acc);
  }
  int n = nt * 16 + (lane & 15);
  int hi8 = (lane >> 4) << 3;
  float bv = bias ? bias[n] : 0.0f;
#pragma unroll
  for (int v = 0; v < 8; ++v) {
    int m = mt * 16 + v + hi8;
    float xv = acc[v] + bv;
    if (act == 1) xv = fmaxf(xv, 0.0f);
    else if (act == 2) xv = tanhf(xv);
    if (Yf)  Yf[(size_t)m * ldYf + n] = xv;
    if (Yh)  Yh[(size_t)m * N + n]    = (_Float16)xv;
    if (YhT) YhT[(size_t)n * M + m]   = (_Float16)xv;
  }
}

// ---------------- attention kernel: one block = (batch b, 16-row i-tile) ----------------
__global__ void attn_kernel(const _Float16* __restrict__ q16,
                            const _Float16* __restrict__ k16,
                            const _Float16* __restrict__ vT16,   // [TD][TB*TT]
                            const float*    __restrict__ qpe,    // [TB*TT][PEK_N]
                            const _Float16* __restrict__ pevT,   // [TD][PEV_K]
                            _Float16* __restrict__ c16out) {
  __shared__ __align__(16) float    s_s[16 * TT];        // 64KB f32 scores
  __shared__ __align__(16) _Float16 s_a16[16 * TT];      // 32KB f16 unnormalized exp (WMMA A source)
  __shared__ __align__(16) _Float16 s_aw[16 * PEV_K];    // bucketed (unnormalized) attention sums
  __shared__ float s_red[4][256];
  __shared__ float s_l[16];

  int b    = blockIdx.x >> 6;
  int i0   = (blockIdx.x & 63) * 16;
  int row0 = b * TT + i0;
  int wave = threadIdx.x >> 5;
  int lane = threadIdx.x & 31;
  int nl   = lane & 15;
  int hi8  = (lane >> 4) << 3;

  // preload the 16x128 q tile as 4 A-chunks (reused across all j-tiles of this wave)
  v16h qa[4];
#pragma unroll
  for (int kk = 0; kk < 4; ++kk)
    qa[kk] = ld_a16(q16 + (size_t)row0 * TD + kk * 32, TD);

  // ---- phase 1: scores s[i,j] = q.k + qpe[i, clamp(K+j-i)] ----
  for (int jt = wave; jt < TT / 16; jt += 8) {
    int j0 = jt * 16;
    v8f acc = {};
#pragma unroll
    for (int kk = 0; kk < 4; ++kk) {
      v16h bm = ld_b16(k16 + (size_t)(b * TT + j0) * TD + kk * 32, TD);
      acc = wmma_f16(qa[kk], bm, acc);
    }
    int j = j0 + nl;
#pragma unroll
    for (int v = 0; v < 8; ++v) {
      int m = v + hi8;
      int ig = i0 + m;
      int id = RK + j - ig;
      id = id < 0 ? 0 : (id > 2 * RK ? 2 * RK : id);
      s_s[m * TT + j] = acc[v] + qpe[(size_t)(row0 + m) * PEK_N + id];
    }
  }
  __syncthreads();

  // ---- phase 2: softmax (unnormalized exp kept; normalize by l at the end) ----
  int i = threadIdx.x >> 4;
  int c = threadIdx.x & 15;
  int ig = i0 + i;
  float mx = -3.0e38f;
  for (int j = c; j < TT; j += 16) mx = fmaxf(mx, s_s[i * TT + j]);
  s_red[0][threadIdx.x] = mx;
  __syncthreads();
  float rowmax = -3.0e38f;
#pragma unroll
  for (int t = 0; t < 16; ++t) rowmax = fmaxf(rowmax, s_red[0][i * 16 + t]);

  float sum = 0.f, e0 = 0.f, e1 = 0.f;
  for (int j = c; j < TT; j += 16) {
    float ev = __expf(s_s[i * TT + j] - rowmax);
    sum += ev;
    if (j <= ig - RK) e0 += ev;           // rel-id clamped to 0
    if (j >= ig + RK) e1 += ev;           // rel-id clamped to 2K
    s_a16[i * TT + j] = (_Float16)ev;
  }
  s_red[1][threadIdx.x] = sum;
  s_red[2][threadIdx.x] = e0;
  s_red[3][threadIdx.x] = e1;
  __syncthreads();

  if (c == 0) {
    float l = 0.f, a0 = 0.f, a1 = 0.f;
#pragma unroll
    for (int t = 0; t < 16; ++t) {
      l  += s_red[1][i * 16 + t];
      a0 += s_red[2][i * 16 + t];
      a1 += s_red[3][i * 16 + t];
    }
    s_l[i] = l;
    s_aw[i * PEV_K + 0]      = (_Float16)a0;
    s_aw[i * PEV_K + 2 * RK] = (_Float16)a1;
  }
  // middle buckets r=1..2K-1: exactly one j = ig + r - K; pad r in [129,160) with zero
  for (int r = c; r < PEV_K; r += 16) {
    if (r == 0 || r == 2 * RK) continue;
    float val = 0.f;
    if (r < 2 * RK) {
      int j = ig + r - RK;
      if (j >= 0 && j < TT) val = (float)s_a16[i * TT + j];
    }
    s_aw[i * PEV_K + r] = (_Float16)val;
  }
  __syncthreads();

  // ---- phase 3: c = (e @ V + aw @ peV^T) / l ; each wave owns one 16-wide d-tile ----
  int n0 = wave * 16;
  v8f acc = {};
  for (int kk = 0; kk < TT; kk += 32) {
    v16h a  = ld_a16(s_a16 + kk, TT);
    v16h bm = ld_b16(vT16 + (size_t)n0 * (TB * TT) + b * TT + kk, TB * TT);
    acc = wmma_f16(a, bm, acc);
  }
#pragma unroll
  for (int kk = 0; kk < PEV_K; kk += 32) {
    v16h a  = ld_a16(s_aw + kk, PEV_K);
    v16h bm = ld_b16(pevT + n0 * PEV_K + kk, PEV_K);
    acc = wmma_f16(a, bm, acc);
  }
#pragma unroll
  for (int v = 0; v < 8; ++v) {
    int m = v + hi8;
    c16out[(size_t)(row0 + m) * TD + n0 + nl] = (_Float16)(acc[v] / s_l[m]);
  }
}

// ---------------- host ----------------
extern "C" void kernel_launch(void* const* d_in, const int* in_sizes, int n_in,
                              void* d_out, int out_size, void* d_ws, size_t ws_size,
                              hipStream_t stream) {
  (void)in_sizes; (void)n_in; (void)out_size; (void)ws_size;
  const float* x   = (const float*)d_in[0];
  const float* W0  = (const float*)d_in[1];
  const float* b0  = (const float*)d_in[2];
  const float* Wq  = (const float*)d_in[3];
  const float* bq  = (const float*)d_in[4];
  const float* Wk  = (const float*)d_in[5];
  const float* bk  = (const float*)d_in[6];
  const float* Wv  = (const float*)d_in[7];
  const float* bv  = (const float*)d_in[8];
  const float* W1  = (const float*)d_in[9];
  const float* b1  = (const float*)d_in[10];
  const float* pek = (const float*)d_in[11];
  const float* pev = (const float*)d_in[12];

  const int M = TB * TT;                     // 8192 rows
  char* w = (char*)d_ws;
  size_t off = 0;
  auto alloc = [&](size_t bytes) -> void* {
    void* p = w + off;
    off = (off + bytes + 255) & ~(size_t)255;
    return p;
  };
  _Float16* x16  = (_Float16*)alloc((size_t)M * TD * 2);
  _Float16* W0T  = (_Float16*)alloc(TD * TD * 2);
  _Float16* WqT  = (_Float16*)alloc(TD * TD * 2);
  _Float16* WkT  = (_Float16*)alloc(TD * TD * 2);
  _Float16* WvT  = (_Float16*)alloc(TD * TD * 2);
  _Float16* W1T  = (_Float16*)alloc(TD * TD * 2);
  _Float16* pekp = (_Float16*)alloc(PEK_N * TD * 2);
  _Float16* pevT = (_Float16*)alloc(TD * PEV_K * 2);
  _Float16* hs16 = (_Float16*)alloc((size_t)M * TD * 2);
  _Float16* q16  = (_Float16*)alloc((size_t)M * TD * 2);
  _Float16* k16  = (_Float16*)alloc((size_t)M * TD * 2);
  _Float16* vT16 = (_Float16*)alloc((size_t)M * TD * 2);
  float*    qpe  = (float*)   alloc((size_t)M * PEK_N * 4);
  _Float16* c16  = (_Float16*)alloc((size_t)M * TD * 2);

  // prep
  cvt_f16_kernel<<<(M * TD) / 256, 256, 0, stream>>>(x, x16, M * TD);
  transpose_w_kernel<<<64, 256, 0, stream>>>(W0, W0T);
  transpose_w_kernel<<<64, 256, 0, stream>>>(Wq, WqT);
  transpose_w_kernel<<<64, 256, 0, stream>>>(Wk, WkT);
  transpose_w_kernel<<<64, 256, 0, stream>>>(Wv, WvT);
  transpose_w_kernel<<<64, 256, 0, stream>>>(W1, W1T);
  prep_pek_kernel<<<(PEK_N * TD + 255) / 256, 256, 0, stream>>>(pek, pekp);
  prep_pev_kernel<<<(TD * PEV_K + 255) / 256, 256, 0, stream>>>(pev, pevT);

  // hs = relu(x @ W0 + b0)
  gemm16_kernel<<<512, 256, 0, stream>>>(x16, W0T, b0, nullptr, hs16, nullptr, M, TD, TD, 1, 0);
  // q/k/v = tanh(hs @ W + b); v stored transposed for the AV matmul
  gemm16_kernel<<<512, 256, 0, stream>>>(hs16, WqT, bq, nullptr, q16, nullptr, M, TD, TD, 2, 0);
  gemm16_kernel<<<512, 256, 0, stream>>>(hs16, WkT, bk, nullptr, k16, nullptr, M, TD, TD, 2, 0);
  gemm16_kernel<<<512, 256, 0, stream>>>(hs16, WvT, bv, nullptr, nullptr, vT16, M, TD, TD, 2, 0);
  // qpe = q @ pe_key^T  (N padded 129->144)
  gemm16_kernel<<<(M / 16 * (PEK_N / 16) + 7) / 8, 256, 0, stream>>>(
      q16, pekp, nullptr, qpe, nullptr, nullptr, M, PEK_N, TD, 0, PEK_N);
  // attention + RPR value term -> c (f16)
  attn_kernel<<<TB * (TT / 16), 256, 0, stream>>>(q16, k16, vT16, qpe, pevT, c16);
  // out = relu(c @ W1 + b1) -> f32
  gemm16_kernel<<<512, 256, 0, stream>>>(c16, W1T, b1, (float*)d_out, nullptr, nullptr, M, TD, TD, 1, TD);
}
